// WrappedEquiformer_16312285790694
// MI455X (gfx1250) — matrix operand order repaired and checked by reading, not compiled, not run on hardware
//
#include <hip/hip_runtime.h>
#include <cstdint>
#include <cstddef>

// ---------------- problem constants ----------------
constexpr int NTOK = 2048;
constexpr int D0c = 256, D1c = 128, D2c = 32;
constexpr int Hc = 4, DHc = 64;
constexpr int Fc = 416;          // D0 + D1 + D2
constexpr int Lc = 4;
constexpr int NCc = 10;
constexpr int FFc = 512;         // 2*D0
constexpr size_t NNp = (size_t)NTOK * NTOK;

// ---------------- gfx1250 async global->LDS availability ----------------
#if defined(__HIP_DEVICE_COMPILE__) && defined(__gfx1250__) && \
    __has_builtin(__builtin_amdgcn_global_load_async_to_lds_b128) && \
    __has_builtin(__builtin_amdgcn_s_wait_asynccnt)
#define HAVE_ASYNC_LDS 1
#else
#define HAVE_ASYNC_LDS 0
#endif

#if HAVE_ASYNC_LDS
typedef int v4i_vs __attribute__((vector_size(16)));
typedef __attribute__((address_space(1))) v4i_vs* gv4i_ptr;
typedef __attribute__((address_space(3))) v4i_vs* lv4i_ptr;
#endif

// ---------------- bf16 helpers ----------------
typedef __attribute__((ext_vector_type(16))) __bf16 bf16x16;
typedef __attribute__((ext_vector_type(8)))  float  floatx8;

__device__ __forceinline__ unsigned short f2bf(float f) {
  unsigned int u = __builtin_bit_cast(unsigned int, f);
  u += 0x7FFFu + ((u >> 16) & 1u);           // round-to-nearest-even
  return (unsigned short)(u >> 16);
}
__device__ __forceinline__ float bf2f(unsigned short s) {
  unsigned int u = ((unsigned int)s) << 16;
  return __builtin_bit_cast(float, u);
}

struct alignas(16) U4 { unsigned int a, b, c, d; };

union FragAB {
  bf16x16 v;
  unsigned short us[16];
  U4 q[2];
};
union Chunk8 {
  U4 q;
  unsigned short us[8];
};

// ---------------- generic bf16 WMMA GEMM with LDS staging ----------------
// C[M,Nc](f32) = op( (A ⊙ A2?)[M,K](bf16) @ B[K,Nc](bf16) + bias? ) [+ C if ACC]
//
// Block = 256 threads = 8 waves; block tile 128x64, wave grid 4x2, each wave
// owns a 32x32 C tile (2x2 WMMA fragments), K stepped by 32.
// Staging: A tile 128x32 (stride 40), B tile 32x64 (stride 72) in LDS,
// zero-filled past Nc so the WMMA main loop has NO divergent control flow
// (EXEC stays all-1s as the ISA requires). A fragment reads are 16B ds loads, B reads are 2-byte gathers
// following the CDNA5 WMMA VGPR layouts (CDNA5 ISA §7.12.2):
//   A 16x32: lane r/half: us[0..7]=K(8*half+e), us[8..15]=K(16+8*half+e)
//   B 32x16: lane r = N col, us[0..7]=K(8*half+e), us[8..15]=K(16+8*half+e)
//   C 16x16: acc[e] at lane -> row = e+8*half, col = r
constexpr int BM = 128, BN = 64, BK = 32;
constexpr int LDA_SH = 40;   // padded LDS stride (elements) for A tile
constexpr int LDB_SH = 72;   // padded LDS stride (elements) for B tile

template <bool MUL, bool ACC, int OP, bool BIAS>   // OP: 0 none, 1 silu, 2 sigmoid
__global__ __launch_bounds__(256) void gemm_bf16_kernel(
    const unsigned short* __restrict__ A,
    const unsigned short* __restrict__ A2,
    const unsigned short* __restrict__ B,
    float* __restrict__ C,
    const float* __restrict__ bias,
    int M, int Nc, int K, int lda, int ldb, int ldc)
{
  __shared__ unsigned short Ash[BM * LDA_SH];
  __shared__ unsigned short Bsh[BK * LDB_SH];

  const int tid  = threadIdx.x;
  const int lane = tid & 31;
  const int wave = __builtin_amdgcn_readfirstlane(tid) >> 5;  // uniform 0..7
  const int r    = lane & 15;
  const int half = lane >> 4;
  const int wrow = wave >> 1;                                 // 0..3
  const int wcol = wave & 1;                                  // 0..1
  const int blockRow = blockIdx.y * BM;
  const int blockCol = blockIdx.x * BN;

  // staging work assignment
  const int aRow0 = tid >> 2;          // A: 2 chunks of 8 bf16 per thread
  const int aCol  = (tid & 3) * 8;
  const int bRow  = tid >> 3;          // B: 1 chunk of 8 bf16 per thread
  const int bCol  = (tid & 7) * 8;

  floatx8 acc[2][2] = {};

  for (int k0 = 0; k0 < K; k0 += BK) {
    __syncthreads();                   // previous tile fully consumed

    // ---- stage A (optionally fused with elementwise A2 multiply) ----
#pragma unroll
    for (int s = 0; s < 2; ++s) {
      const int rowp = aRow0 + s * 64;
      const unsigned short* ap = A + (size_t)(blockRow + rowp) * lda + k0 + aCol;
      unsigned short* dst = &Ash[rowp * LDA_SH + aCol];
      if (MUL) {
        const unsigned short* mp = A2 + (size_t)(blockRow + rowp) * lda + k0 + aCol;
        Chunk8 av, mv;
        av.q = *reinterpret_cast<const U4*>(ap);
        mv.q = *reinterpret_cast<const U4*>(mp);
#pragma unroll
        for (int e = 0; e < 8; ++e)
          av.us[e] = f2bf(bf2f(av.us[e]) * bf2f(mv.us[e]));
        *reinterpret_cast<U4*>(dst) = av.q;
      } else {
#if HAVE_ASYNC_LDS
        __builtin_amdgcn_global_load_async_to_lds_b128(
            (gv4i_ptr)(ap), (lv4i_ptr)(dst), 0, 0);
#else
        *reinterpret_cast<U4*>(dst) = *reinterpret_cast<const U4*>(ap);
#endif
      }
    }

    // ---- stage B with zero-fill past Nc ----
    {
      const int gcol = blockCol + bCol;
      const unsigned short* bp = B + (size_t)(k0 + bRow) * ldb + gcol;
      unsigned short* dst = &Bsh[bRow * LDB_SH + bCol];
      if (gcol + 8 <= Nc) {
#if HAVE_ASYNC_LDS
        __builtin_amdgcn_global_load_async_to_lds_b128(
            (gv4i_ptr)(bp), (lv4i_ptr)(dst), 0, 0);
#else
        *reinterpret_cast<U4*>(dst) = *reinterpret_cast<const U4*>(bp);
#endif
      } else {
        Chunk8 bv;
#pragma unroll
        for (int e = 0; e < 8; ++e)
          bv.us[e] = (gcol + e < Nc) ? bp[e] : (unsigned short)0;
        *reinterpret_cast<U4*>(dst) = bv.q;
      }
    }
#if HAVE_ASYNC_LDS
    __builtin_amdgcn_s_wait_asynccnt(0);
#endif
    __syncthreads();

    // ---- fragment loads (A: ds_load_b128, B: 2-byte gathers) + WMMA, no divergence ----
    FragAB a[2], b[2];
#pragma unroll
    for (int i = 0; i < 2; ++i) {
      const unsigned short* ap = &Ash[(wrow * 32 + i * 16 + r) * LDA_SH + 8 * half];
      a[i].q[0] = *reinterpret_cast<const U4*>(ap);
      a[i].q[1] = *reinterpret_cast<const U4*>(ap + 16);
    }
#pragma unroll
    for (int j = 0; j < 2; ++j) {
      const unsigned short* bp = &Bsh[wcol * 32 + j * 16 + r];
#pragma unroll
      for (int e = 0; e < 8; ++e) {
        b[j].us[e] = bp[(8 * half + e) * LDB_SH];
        b[j].us[8 + e] = bp[(16 + 8 * half + e) * LDB_SH];
      }
    }
#pragma unroll
    for (int i = 0; i < 2; ++i)
#pragma unroll
      for (int j = 0; j < 2; ++j)
        acc[i][j] = __builtin_amdgcn_wmma_f32_16x16x32_bf16(
            false, a[i].v, false, b[j].v, (short)0, acc[i][j], false, false);
  }

  // ---- straight-line epilogue (only stores are lane-masked) ----
#pragma unroll
  for (int i = 0; i < 2; ++i) {
#pragma unroll
    for (int j = 0; j < 2; ++j) {
      const int col = blockCol + wcol * 32 + j * 16 + r;
      if (col < Nc) {
        const float bv = BIAS ? bias[col] : 0.0f;
#pragma unroll
        for (int e = 0; e < 8; ++e) {
          const int row = blockRow + wrow * 32 + i * 16 + e + 8 * half;
          float v = acc[i][j][e] + bv;
          if (OP == 1) v = v * (1.0f / (1.0f + __expf(-v)));   // SiLU
          if (OP == 2) v = 1.0f / (1.0f + __expf(-v));         // sigmoid
          const size_t idx = (size_t)row * ldc + col;
          if (ACC) v += C[idx];
          C[idx] = v;
        }
      }
    }
  }
}

// ---------------- small helper kernels ----------------
__global__ void cvt_bf16_kernel(const float* __restrict__ s,
                                unsigned short* __restrict__ d, size_t n) {
  size_t i = (size_t)blockIdx.x * 256 + threadIdx.x;
  if (i < n) d[i] = f2bf(s[i]);
}

__global__ void cvt_padcols_kernel(const float* __restrict__ s,
                                   unsigned short* __restrict__ d,
                                   int rows, int sc, int dc) {
  int i = blockIdx.x * 256 + threadIdx.x;
  if (i >= rows * dc) return;
  int rr = i / dc, c = i % dc;
  d[i] = (c < sc) ? f2bf(s[(size_t)rr * sc + c]) : (unsigned short)0;
}

__global__ void cvt_padrows_kernel(const float* __restrict__ s,
                                   unsigned short* __restrict__ d,
                                   int dr, int sr, int cols) {
  int i = blockIdx.x * 256 + threadIdx.x;
  if (i >= dr * cols) return;
  int rr = i / cols;
  d[i] = (rr < sr) ? f2bf(s[i]) : (unsigned short)0;
}

// kT[(h*DH+dd)*N + m] = kf[m*256 + h*DH + dd]
__global__ void kT_kernel(const float* __restrict__ kf,
                          unsigned short* __restrict__ kT) {
  int i = blockIdx.x * 256 + threadIdx.x;
  if (i >= Hc * DHc * NTOK) return;
  int m = i % NTOK, hd = i / NTOK;
  kT[i] = f2bf(kf[(size_t)m * (Hc * DHc) + hd]);
}

// pairwise real spherical harmonics -> bf16 component-major planes
__global__ void sph_kernel(const float* __restrict__ pos,
                           unsigned short* __restrict__ Y1,
                           unsigned short* __restrict__ Y2) {
  int idx = blockIdx.x * 256 + threadIdx.x;
  if (idx >= (int)NNp) return;
  int i = idx / NTOK, j = idx % NTOK;
  float rx = pos[j * 3 + 0] - pos[i * 3 + 0];
  float ry = pos[j * 3 + 1] - pos[i * 3 + 1];
  float rz = pos[j * 3 + 2] - pos[i * 3 + 2];
  float dist = sqrtf(rx * rx + ry * ry + rz * rz);
  float valid = (dist > 1e-6f) ? 1.0f : 0.0f;
  float inv = valid / fmaxf(dist, 1e-6f);
  float ux = rx * inv, uy = ry * inv, uz = rz * inv;
  const float C2v = 1.7320508075688772f;
  Y1[0 * NNp + idx] = f2bf(ux);
  Y1[1 * NNp + idx] = f2bf(uy);
  Y1[2 * NNp + idx] = f2bf(uz);
  Y2[0 * NNp + idx] = f2bf(C2v * ux * uy * valid);
  Y2[1 * NNp + idx] = f2bf(C2v * uy * uz * valid);
  Y2[2 * NNp + idx] = f2bf(0.5f * (3.0f * uz * uz - 1.0f) * valid);
  Y2[3 * NNp + idx] = f2bf(C2v * ux * uz * valid);
  Y2[4 * NNp + idx] = f2bf(0.5f * C2v * (ux * ux - uy * uy) * valid);
}

// inv[n, 0:256]=x0 ; [256:384]=||x1||_c ; [384:416]=||x2||_c  (bf16)
__global__ void invariants_kernel(const float* __restrict__ x0,
                                  const float* __restrict__ x1p,
                                  const float* __restrict__ x2p,
                                  unsigned short* __restrict__ inv) {
  int n = blockIdx.x, t = threadIdx.x;
  inv[(size_t)n * Fc + t] = f2bf(x0[(size_t)n * D0c + t]);
  if (t < D1c) {
    float s = 0.f;
#pragma unroll
    for (int c = 0; c < 3; ++c) {
      float v = x1p[(size_t)c * NTOK * D1c + (size_t)n * D1c + t];
      s += v * v;
    }
    inv[(size_t)n * Fc + D0c + t] = f2bf(sqrtf(s));
  }
  if (t < D2c) {
    float s = 0.f;
#pragma unroll
    for (int c = 0; c < 5; ++c) {
      float v = x2p[(size_t)c * NTOK * D2c + (size_t)n * D2c + t];
      s += v * v;
    }
    inv[(size_t)n * Fc + D0c + D1c + t] = f2bf(sqrtf(s));
  }
}

// row softmax (f32 scores -> bf16 probs), scale 1/sqrt(DH)=0.125 applied
__global__ __launch_bounds__(256) void softmax_kernel(
    const float* __restrict__ S, unsigned short* __restrict__ P) {
  int row = blockIdx.x, h = blockIdx.y, t = threadIdx.x;
  const float* s = S + ((size_t)h * NTOK + row) * NTOK;
  unsigned short* p = P + ((size_t)h * NTOK + row) * NTOK;
  float v[8];
  float mx = -3.4e38f;
#pragma unroll
  for (int k = 0; k < 8; ++k) { v[k] = s[t + k * 256]; mx = fmaxf(mx, v[k]); }
  __shared__ float red[256];
  red[t] = mx; __syncthreads();
  for (int off = 128; off > 0; off >>= 1) {
    if (t < off) red[t] = fmaxf(red[t], red[t + off]);
    __syncthreads();
  }
  mx = red[0]; __syncthreads();
  float sum = 0.f;
#pragma unroll
  for (int k = 0; k < 8; ++k) { v[k] = __expf((v[k] - mx) * 0.125f); sum += v[k]; }
  red[t] = sum; __syncthreads();
  for (int off = 128; off > 0; off >>= 1) {
    if (t < off) red[t] += red[t + off];
    __syncthreads();
  }
  float rs = 1.0f / red[0];
#pragma unroll
  for (int k = 0; k < 8; ++k) p[t + k * 256] = f2bf(v[k] * rs);
}

__global__ void mean_attn_kernel(const unsigned short* __restrict__ attn,
                                 unsigned short* __restrict__ ab) {
  size_t i = (size_t)blockIdx.x * 256 + threadIdx.x;
  if (i >= NNp) return;
  float s = bf2f(attn[i]) + bf2f(attn[NNp + i]) +
            bf2f(attn[2 * NNp + i]) + bf2f(attn[3 * NNp + i]);
  ab[i] = f2bf(s * 0.25f);
}

// x1 *= g[:, :D1, None] ; x2 *= g[:, D1:, None]
__global__ void gate_kernel(const float* __restrict__ g,
                            float* __restrict__ x1p, float* __restrict__ x2p) {
  int n = blockIdx.x, t = threadIdx.x;
  if (t < D1c) {
    float gv = g[(size_t)n * (D1c + D2c) + t];
#pragma unroll
    for (int c = 0; c < 3; ++c)
      x1p[(size_t)c * NTOK * D1c + (size_t)n * D1c + t] *= gv;
  } else if (t < D1c + D2c) {
    int dd = t - D1c;
    float gv = g[(size_t)n * (D1c + D2c) + t];
#pragma unroll
    for (int c = 0; c < 5; ++c)
      x2p[(size_t)c * NTOK * D2c + (size_t)n * D2c + dd] *= gv;
  }
}

// ---------------- host-side launch helpers ----------------
static inline void launch_gemm(hipStream_t s, const unsigned short* A,
                               const unsigned short* A2, const unsigned short* B,
                               float* C, const float* bias, int M, int Nc, int K,
                               int lda, int ldb, int ldc, int acc, int op) {
  dim3 grid((Nc + BN - 1) / BN, M / BM), blk(256);
  if (A2)
    gemm_bf16_kernel<true, true, 0, false><<<grid, blk, 0, s>>>(A, A2, B, C, bias, M, Nc, K, lda, ldb, ldc);
  else if (bias)
    gemm_bf16_kernel<false, false, 0, true><<<grid, blk, 0, s>>>(A, A2, B, C, bias, M, Nc, K, lda, ldb, ldc);
  else if (op == 1)
    gemm_bf16_kernel<false, false, 1, false><<<grid, blk, 0, s>>>(A, A2, B, C, bias, M, Nc, K, lda, ldb, ldc);
  else if (op == 2)
    gemm_bf16_kernel<false, false, 2, false><<<grid, blk, 0, s>>>(A, A2, B, C, bias, M, Nc, K, lda, ldb, ldc);
  else if (acc)
    gemm_bf16_kernel<false, true, 0, false><<<grid, blk, 0, s>>>(A, A2, B, C, bias, M, Nc, K, lda, ldb, ldc);
  else
    gemm_bf16_kernel<false, false, 0, false><<<grid, blk, 0, s>>>(A, A2, B, C, bias, M, Nc, K, lda, ldb, ldc);
}

static inline void launch_cvt(hipStream_t s, const float* src, unsigned short* dst, size_t n) {
  cvt_bf16_kernel<<<(unsigned)((n + 255) / 256), 256, 0, s>>>(src, dst, n);
}

extern "C" void kernel_launch(void* const* d_in, const int* in_sizes, int n_in,
                              void* d_out, int out_size, void* d_ws, size_t ws_size,
                              hipStream_t stream) {
  (void)in_sizes; (void)n_in; (void)out_size; (void)ws_size;

  const float* x     = (const float*)d_in[0];
  const float* pos   = (const float*)d_in[1];
  const float* embW  = (const float*)d_in[2];
  const float* embB  = (const float*)d_in[3];
  const float* Wq    = (const float*)d_in[4];
  const float* Wk    = (const float*)d_in[5];
  const float* Wv0   = (const float*)d_in[6];
  const float* Wv1   = (const float*)d_in[7];
  const float* Wv2   = (const float*)d_in[8];
  const float* Ws1   = (const float*)d_in[9];
  const float* Ws2   = (const float*)d_in[10];
  const float* Wffa  = (const float*)d_in[11];
  const float* Wffb  = (const float*)d_in[12];
  const float* Wg    = (const float*)d_in[13];
  const float* outW  = (const float*)d_in[14];
  const float* outB  = (const float*)d_in[15];

  // ---- workspace bump allocator (256B aligned) ----
  size_t off = 0;
  auto alloc = [&](size_t bytes) -> void* {
    void* p = (char*)d_ws + off;
    off += (bytes + 255) & ~(size_t)255;
    return p;
  };
  unsigned short* Y1b   = (unsigned short*)alloc(3 * NNp * 2);
  unsigned short* Y2b   = (unsigned short*)alloc(5 * NNp * 2);
  float*          scores= (float*)alloc(4 * NNp * 4);
  unsigned short* attnb = (unsigned short*)alloc(4 * NNp * 2);
  unsigned short* ab    = (unsigned short*)alloc(NNp * 2);
  float*          x0    = (float*)alloc((size_t)NTOK * D0c * 4);
  unsigned short* x0b   = (unsigned short*)alloc((size_t)NTOK * D0c * 2);
  float*          x1p   = (float*)alloc((size_t)3 * NTOK * D1c * 4);
  unsigned short* x1b   = (unsigned short*)alloc((size_t)3 * NTOK * D1c * 2);
  float*          x2p   = (float*)alloc((size_t)5 * NTOK * D2c * 4);
  unsigned short* x2b   = (unsigned short*)alloc((size_t)5 * NTOK * D2c * 2);
  unsigned short* invb  = (unsigned short*)alloc((size_t)NTOK * Fc * 2);
  float*          qf    = (float*)alloc((size_t)NTOK * 256 * 4);
  unsigned short* qb    = (unsigned short*)alloc((size_t)NTOK * 256 * 2);
  float*          kf    = (float*)alloc((size_t)NTOK * 256 * 4);
  unsigned short* kTb   = (unsigned short*)alloc((size_t)256 * NTOK * 2);
  float*          v0f   = (float*)alloc((size_t)NTOK * 256 * 4);
  unsigned short* v0b   = (unsigned short*)alloc((size_t)NTOK * 256 * 2);
  float*          v1f   = (float*)alloc((size_t)3 * NTOK * D1c * 4);
  unsigned short* v1b   = (unsigned short*)alloc((size_t)3 * NTOK * D1c * 2);
  float*          s1f   = (float*)alloc((size_t)NTOK * D1c * 4);
  unsigned short* s1b   = (unsigned short*)alloc((size_t)NTOK * D1c * 2);
  float*          v2f   = (float*)alloc((size_t)5 * NTOK * D2c * 4);
  unsigned short* v2b   = (unsigned short*)alloc((size_t)5 * NTOK * D2c * 2);
  float*          s2f   = (float*)alloc((size_t)NTOK * D2c * 4);
  unsigned short* s2b   = (unsigned short*)alloc((size_t)NTOK * D2c * 2);
  float*          ff1f  = (float*)alloc((size_t)NTOK * FFc * 4);
  unsigned short* ff1b  = (unsigned short*)alloc((size_t)NTOK * FFc * 2);
  float*          gf    = (float*)alloc((size_t)NTOK * (D1c + D2c) * 4);
  unsigned short* Wqb   = (unsigned short*)alloc((size_t)Lc * Fc * 256 * 2);
  unsigned short* Wkb   = (unsigned short*)alloc((size_t)Lc * Fc * 256 * 2);
  unsigned short* Wv0b  = (unsigned short*)alloc((size_t)Lc * 256 * 256 * 2);
  unsigned short* Wv1b  = (unsigned short*)alloc((size_t)Lc * D1c * D1c * 2);
  unsigned short* Wv2b  = (unsigned short*)alloc((size_t)Lc * D2c * D2c * 2);
  unsigned short* Ws1b  = (unsigned short*)alloc((size_t)Lc * 256 * D1c * 2);
  unsigned short* Ws2b  = (unsigned short*)alloc((size_t)Lc * 256 * D2c * 2);
  unsigned short* Wffab = (unsigned short*)alloc((size_t)Lc * 256 * FFc * 2);
  unsigned short* Wffbb = (unsigned short*)alloc((size_t)Lc * FFc * 256 * 2);
  unsigned short* Wgb   = (unsigned short*)alloc((size_t)Lc * Fc * (D1c + D2c) * 2);
  unsigned short* embWb = (unsigned short*)alloc((size_t)128 * 256 * 2);
  unsigned short* outWb = (unsigned short*)alloc((size_t)256 * 16 * 2);
  unsigned short* xpb   = (unsigned short*)alloc((size_t)NTOK * 128 * 2);

  // ---- one-time prep ----
  sph_kernel<<<(unsigned)(NNp / 256), 256, 0, stream>>>(pos, Y1b, Y2b);
  launch_cvt(stream, Wq,   Wqb,   (size_t)Lc * Fc * 256);
  launch_cvt(stream, Wk,   Wkb,   (size_t)Lc * Fc * 256);
  launch_cvt(stream, Wv0,  Wv0b,  (size_t)Lc * 256 * 256);
  launch_cvt(stream, Wv1,  Wv1b,  (size_t)Lc * D1c * D1c);
  launch_cvt(stream, Wv2,  Wv2b,  (size_t)Lc * D2c * D2c);
  launch_cvt(stream, Ws1,  Ws1b,  (size_t)Lc * 256 * D1c);
  launch_cvt(stream, Ws2,  Ws2b,  (size_t)Lc * 256 * D2c);
  launch_cvt(stream, Wffa, Wffab, (size_t)Lc * 256 * FFc);
  launch_cvt(stream, Wffb, Wffbb, (size_t)Lc * FFc * 256);
  launch_cvt(stream, Wg,   Wgb,   (size_t)Lc * Fc * (D1c + D2c));
  cvt_padcols_kernel<<<(NTOK * 128 + 255) / 256, 256, 0, stream>>>(x, xpb, NTOK, 118, 128);
  cvt_padrows_kernel<<<(128 * 256 + 255) / 256, 256, 0, stream>>>(embW, embWb, 128, 118, 256);
  cvt_padcols_kernel<<<(256 * 16 + 255) / 256, 256, 0, stream>>>(outW, outWb, 256, 10, 16);

  // x0 = x @ emb_W + emb_b ; x1 = x2 = 0
  launch_gemm(stream, xpb, nullptr, embWb, x0, embB, NTOK, 256, 128, 128, 256, 256, 0, 0);
  (void)hipMemsetAsync(x1p, 0, (size_t)3 * NTOK * D1c * 4, stream);
  (void)hipMemsetAsync(x2p, 0, (size_t)5 * NTOK * D2c * 4, stream);

  for (int l = 0; l < Lc; ++l) {
    invariants_kernel<<<NTOK, 256, 0, stream>>>(x0, x1p, x2p, invb);
    launch_cvt(stream, x0, x0b, (size_t)NTOK * D0c);
    launch_cvt(stream, x1p, x1b, (size_t)3 * NTOK * D1c);
    launch_cvt(stream, x2p, x2b, (size_t)5 * NTOK * D2c);

    // q, k projections
    launch_gemm(stream, invb, nullptr, Wqb + (size_t)l * Fc * 256, qf, nullptr,
                NTOK, 256, Fc, Fc, 256, 256, 0, 0);
    launch_gemm(stream, invb, nullptr, Wkb + (size_t)l * Fc * 256, kf, nullptr,
                NTOK, 256, Fc, Fc, 256, 256, 0, 0);
    launch_cvt(stream, qf, qb, (size_t)NTOK * 256);
    kT_kernel<<<(Hc * DHc * NTOK + 255) / 256, 256, 0, stream>>>(kf, kTb);

    // per-head scores = q_h @ k_h^T, softmax, head-mean
    for (int h = 0; h < Hc; ++h)
      launch_gemm(stream, qb + h * DHc, nullptr, kTb + (size_t)h * DHc * NTOK,
                  scores + (size_t)h * NNp, nullptr,
                  NTOK, NTOK, DHc, 256, NTOK, NTOK, 0, 0);
    softmax_kernel<<<dim3(NTOK, Hc), 256, 0, stream>>>(scores, attnb);
    mean_attn_kernel<<<(unsigned)(NNp / 256), 256, 0, stream>>>(attnb, ab);

    // degree-0 aggregation: x0 += attn_h @ v0_h
    launch_gemm(stream, x0b, nullptr, Wv0b + (size_t)l * 256 * 256, v0f, nullptr,
                NTOK, 256, 256, 256, 256, 256, 0, 0);
    launch_cvt(stream, v0f, v0b, (size_t)NTOK * 256);
    for (int h = 0; h < Hc; ++h)
      launch_gemm(stream, attnb + (size_t)h * NNp, nullptr, v0b + h * DHc,
                  x0 + h * DHc, nullptr, NTOK, DHc, NTOK, NTOK, 256, 256, 1, 0);

    // degree-1: v1, s1; x1 += a@v1_c + (a*Y1_c)@s1
    for (int c = 0; c < 3; ++c)
      launch_gemm(stream, x1b + (size_t)c * NTOK * D1c, nullptr,
                  Wv1b + (size_t)l * D1c * D1c, v1f + (size_t)c * NTOK * D1c,
                  nullptr, NTOK, D1c, D1c, D1c, D1c, D1c, 0, 0);
    launch_cvt(stream, v1f, v1b, (size_t)3 * NTOK * D1c);
    launch_gemm(stream, x0b, nullptr, Ws1b + (size_t)l * 256 * D1c, s1f, nullptr,
                NTOK, D1c, 256, 256, D1c, D1c, 0, 0);
    launch_cvt(stream, s1f, s1b, (size_t)NTOK * D1c);
    launch_gemm(stream, x0b, nullptr, Ws2b + (size_t)l * 256 * D2c, s2f, nullptr,
                NTOK, D2c, 256, 256, D2c, D2c, 0, 0);
    launch_cvt(stream, s2f, s2b, (size_t)NTOK * D2c);
    for (int c = 0; c < 3; ++c)
      launch_gemm(stream, ab, nullptr, v1b + (size_t)c * NTOK * D1c,
                  x1p + (size_t)c * NTOK * D1c, nullptr,
                  NTOK, D1c, NTOK, NTOK, D1c, D1c, 1, 0);
    for (int c = 0; c < 3; ++c)
      launch_gemm(stream, ab, Y1b + (size_t)c * NNp, s1b,
                  x1p + (size_t)c * NTOK * D1c, nullptr,
                  NTOK, D1c, NTOK, NTOK, D1c, D1c, 1, 0);

    // degree-2: v2; x2 += a@v2_c + (a*Y2_c)@s2
    for (int c = 0; c < 5; ++c)
      launch_gemm(stream, x2b + (size_t)c * NTOK * D2c, nullptr,
                  Wv2b + (size_t)l * D2c * D2c, v2f + (size_t)c * NTOK * D2c,
                  nullptr, NTOK, D2c, D2c, D2c, D2c, D2c, 0, 0);
    launch_cvt(stream, v2f, v2b, (size_t)5 * NTOK * D2c);
    for (int c = 0; c < 5; ++c)
      launch_gemm(stream, ab, nullptr, v2b + (size_t)c * NTOK * D2c,
                  x2p + (size_t)c * NTOK * D2c, nullptr,
                  NTOK, D2c, NTOK, NTOK, D2c, D2c, 1, 0);
    for (int c = 0; c < 5; ++c)
      launch_gemm(stream, ab, Y2b + (size_t)c * NNp, s2b,
                  x2p + (size_t)c * NTOK * D2c, nullptr,
                  NTOK, D2c, NTOK, NTOK, D2c, D2c, 1, 0);

    // gated feedforward on x0 (post-agg0)
    launch_cvt(stream, x0, x0b, (size_t)NTOK * D0c);
    launch_gemm(stream, x0b, nullptr, Wffab + (size_t)l * 256 * FFc, ff1f, nullptr,
                NTOK, FFc, 256, 256, FFc, FFc, 0, 1 /*silu*/);
    launch_cvt(stream, ff1f, ff1b, (size_t)NTOK * FFc);
    launch_gemm(stream, ff1b, nullptr, Wffbb + (size_t)l * FFc * 256, x0, nullptr,
                NTOK, 256, FFc, FFc, 256, 256, 1, 0);

    // gates
    invariants_kernel<<<NTOK, 256, 0, stream>>>(x0, x1p, x2p, invb);
    launch_gemm(stream, invb, nullptr, Wgb + (size_t)l * Fc * (D1c + D2c), gf,
                nullptr, NTOK, D1c + D2c, Fc, Fc, D1c + D2c, D1c + D2c, 0, 2 /*sigmoid*/);
    gate_kernel<<<NTOK, 256, 0, stream>>>(gf, x1p, x2p);
  }

  // output head: out = x0 @ out_W + out_b
  launch_cvt(stream, x0, x0b, (size_t)NTOK * D0c);
  launch_gemm(stream, x0b, nullptr, outWb, (float*)d_out, outB,
              NTOK, NCc, 256, 256, 16, NCc, 0, 0);
}
